// VAE_Hist_GLM_20856361189425
// MI455X (gfx1250) — matrix operand order repaired and checked by reading, not compiled, not run on hardware
//
#include <hip/hip_runtime.h>
#include <hip/hip_bf16.h>
#include <math.h>

// ---------------------------------------------------------------------------
// VAE_Hist_GLM for MI455X (gfx1250, wave32)
//
// Stage 1: prep      — cos basis -> e/i/hist kernels (10x100 each) + out_filters
// Stage 2: agg(WMMA) — syn_e = S_e @ C_syn_e^T, syn_i = S_i @ C_syn_i^T
//                      f16 WMMA (exact: inputs are {0,1}), f32 accumulate.
//                      Compile-time E -> guard-free main K-loop of
//                      global_load_b128s, explicit two-buffer ping-pong
//                      (no register copies), branch-free clamped tail.
// Stage 3: conv      — depthwise causal filter (len 100), kernels in LDS.
// Stage 4: scan      — single wave, lanes 0..9 = subunits, 128-entry output
//                      ring in LDS, hist taps fully unrolled (coeffs in VGPRs),
//                      double-buffered prefetch of syn rows.
// ---------------------------------------------------------------------------

#define T_DATA   50000
#define E_NO     2000
#define I_NO     500
#define SUB_NO   10
#define SUBP     16          // padded subunit dim (WMMA N)
#define T_NO     100
#define NBASIS   19
#define PI_F     3.14159265358979323846f

typedef __attribute__((ext_vector_type(16))) _Float16 v16h;
typedef __attribute__((ext_vector_type(8)))  float    v8f;

// -------------------------------- Stage 1 ----------------------------------
__global__ __launch_bounds__(256)
void prep_kernel(const float* __restrict__ W_syn,   // [10][19][2]
                 const float* __restrict__ W_hist,  // [10][19]
                 float* __restrict__ kE, float* __restrict__ kI,
                 float* __restrict__ kH,
                 float* __restrict__ out_filters)   // d_out + 50000, [30][100]
{
    int idx = blockIdx.x * blockDim.x + threadIdx.x;   // (s, tau)
    if (idx >= SUB_NO * T_NO) return;
    int s   = idx / T_NO;
    int tau = idx % T_NO;
    float raw = 5.0f * logf((float)tau + 1.0f);
    float e = 0.f, i = 0.f, h = 0.f;
    #pragma unroll
    for (int b = 0; b < NBASIS; ++b) {
        float phi = 0.5f * PI_F * (float)b;
        float bas = 0.f;
        if (raw >= phi - PI_F && raw <= phi + PI_F)
            bas = 0.5f * cosf(raw - phi) + 0.5f;
        e = fmaf(W_syn[s * 38 + b * 2 + 0], bas, e);
        i = fmaf(W_syn[s * 38 + b * 2 + 1], bas, i);
        h = fmaf(W_hist[s * 19 + b],        bas, h);
    }
    kE[s * T_NO + tau] = e;
    kI[s * T_NO + tau] = i;
    kH[s * T_NO + tau] = h;
    out_filters[(s      ) * T_NO + tau] = e;
    out_filters[(10 + s ) * T_NO + tau] = i;
    out_filters[(20 + s ) * T_NO + tau] = h;
}

// -------------------------------- Stage 2 ----------------------------------
struct Chunk {
    float4 a0, a1, a2, a3;   // A: K = abase+{0..7}, abase+{16..23}
    float4 b0, b1, b2, b3;   // B: K = bbase+{0..7}, bbase+{16..23}
};

__device__ __forceinline__ void load_chunk(Chunk& c,
                                           const float* __restrict__ Arow,
                                           const float* __restrict__ Brow,
                                           int k0, int abase, int bbase)
{
    const float4* pa = (const float4*)(Arow + k0 + abase);
    c.a0 = pa[0]; c.a1 = pa[1];
    const float4* pa2 = (const float4*)(Arow + k0 + abase + 16);
    c.a2 = pa2[0]; c.a3 = pa2[1];
    const float4* pb = (const float4*)(Brow + k0 + bbase);
    c.b0 = pb[0]; c.b1 = pb[1];
    const float4* pb2 = (const float4*)(Brow + k0 + bbase + 16);
    c.b2 = pb2[0]; c.b3 = pb2[1];
}

__device__ __forceinline__ v16h pack16(const float4& x0, const float4& x1,
                                       const float4& x2, const float4& x3)
{
    v16h r;
    r[0]  = (_Float16)x0.x; r[1]  = (_Float16)x0.y;
    r[2]  = (_Float16)x0.z; r[3]  = (_Float16)x0.w;
    r[4]  = (_Float16)x1.x; r[5]  = (_Float16)x1.y;
    r[6]  = (_Float16)x1.z; r[7]  = (_Float16)x1.w;
    r[8]  = (_Float16)x2.x; r[9]  = (_Float16)x2.y;
    r[10] = (_Float16)x2.z; r[11] = (_Float16)x2.w;
    r[12] = (_Float16)x3.x; r[13] = (_Float16)x3.y;
    r[14] = (_Float16)x3.z; r[15] = (_Float16)x3.w;
    return r;
}

__device__ __forceinline__ v16h pack16s(const float4& x0, const float4& x1,
                                        const float4& x2, const float4& x3,
                                        float scale)
{
    v16h r;
    r[0]  = (_Float16)(x0.x * scale); r[1]  = (_Float16)(x0.y * scale);
    r[2]  = (_Float16)(x0.z * scale); r[3]  = (_Float16)(x0.w * scale);
    r[4]  = (_Float16)(x1.x * scale); r[5]  = (_Float16)(x1.y * scale);
    r[6]  = (_Float16)(x1.z * scale); r[7]  = (_Float16)(x1.w * scale);
    r[8]  = (_Float16)(x2.x * scale); r[9]  = (_Float16)(x2.y * scale);
    r[10] = (_Float16)(x2.z * scale); r[11] = (_Float16)(x2.w * scale);
    r[12] = (_Float16)(x3.x * scale); r[13] = (_Float16)(x3.y * scale);
    r[14] = (_Float16)(x3.z * scale); r[15] = (_Float16)(x3.w * scale);
    return r;
}

__device__ __forceinline__ v8f mac_chunk(const Chunk& c, v8f acc, float bscale)
{
    v16h a = pack16 (c.a0, c.a1, c.a2, c.a3);
    v16h b = pack16s(c.b0, c.b1, c.b2, c.b3, bscale);
    return __builtin_amdgcn_wmma_f32_16x16x32_f16(
        false, a, false, b, (short)0, acc, false, false);
}

// One wave computes a 16(time) x 16(sub-padded) tile; K-loop over neurons.
template<int E>
__global__ __launch_bounds__(256)
void agg_wmma_kernel(const float* __restrict__ S,      // [T_DATA][E] spikes
                     const float* __restrict__ C_syn,  // [SUB_NO][E] one-hot
                     float* __restrict__ outP)         // [T_DATA][SUBP]
{
    constexpr int E_MAIN = (E / 32) * 32;              // guard-free chunks
    constexpr int E_PAIR = (E_MAIN / 64) * 64;         // ping-pong pairs
    constexpr int NTILE  = T_DATA / 16;
    const int wave = threadIdx.x >> 5;
    const int lane = threadIdx.x & 31;
    const int tile = blockIdx.x * 8 + wave;
    if (tile >= NTILE) return;

    const int t0    = tile * 16;
    const int m     = lane & 15;           // A row (time) / B column (sub)
    const int abase = (lane >> 4) * 8;     // A K-group: {abase..+7, abase+16..+23}
    const int bbase = (lane >> 4) * 8;     // B K-group: {bbase..+7, bbase+16..+23}
    const float* __restrict__ Arow = S + (size_t)(t0 + m) * E;
    const float* __restrict__ Brow = C_syn + (size_t)(m < SUB_NO ? m : 0) * E;
    const float bscale = (m < SUB_NO) ? 1.0f : 0.0f;   // branch-free N pad

    v8f acc = {};
    Chunk c0, c1;
    load_chunk(c0, Arow, Brow, 0, abase, bbase);

    // explicit ping-pong: wmma(c0) overlaps c1's loads and vice versa;
    // no inter-buffer register copies.
    for (int k0 = 0; k0 < E_PAIR; k0 += 64) {
        load_chunk(c1, Arow, Brow, k0 + 32, abase, bbase);
        acc = mac_chunk(c0, acc, bscale);
        if (k0 + 64 < E_MAIN)
            load_chunk(c0, Arow, Brow, k0 + 64, abase, bbase);
        acc = mac_chunk(c1, acc, bscale);
    }
    if constexpr (E_PAIR < E_MAIN) {       // leftover odd 32-chunk (in c0)
        acc = mac_chunk(c0, acc, bscale);
    }

    // tail chunk (E % 32 != 0): branch-free clamped loads + select-to-zero
    if constexpr (E_MAIN < E) {
        v16h a, b;
        #pragma unroll
        for (int j = 0; j < 16; ++j) {
            int K    = abase + (j < 8 ? j : j + 8);
            int col  = E_MAIN + K;
            int cc   = (col < E) ? col : 0;          // cndmask'd address
            float av = Arow[cc];
            a[j] = (_Float16)((col < E) ? av : 0.f);
            int colb = E_MAIN + bbase + (j < 8 ? j : j + 8);
            int ccb  = (colb < E) ? colb : 0;
            float bv = Brow[ccb];
            b[j] = (_Float16)(((colb < E) ? bv : 0.f) * bscale);
        }
        acc = __builtin_amdgcn_wmma_f32_16x16x32_f16(
            false, a, false, b, (short)0, acc, false, false);
    }

    // D: lane = N (subunit), VGPR v -> M = v + 8*(lane>=16)
    const int ncol  = lane & 15;
    const int mbase = (lane >> 4) * 8;
    #pragma unroll
    for (int v = 0; v < 8; ++v) {
        int row = t0 + mbase + v;
        outP[(size_t)row * SUBP + ncol] = acc[v];
    }
}

// -------------------------------- Stage 3 ----------------------------------
__global__ __launch_bounds__(256)
void conv_kernel(const float* __restrict__ synE,   // [T_DATA][SUBP]
                 const float* __restrict__ synI,
                 const float* __restrict__ kE,     // [10][100]
                 const float* __restrict__ kI,
                 float* __restrict__ syn)          // [T_DATA][SUBP]
{
    __shared__ float se[SUB_NO * T_NO];
    __shared__ float si[SUB_NO * T_NO];
    for (int i = threadIdx.x; i < SUB_NO * T_NO; i += 256) {
        se[i] = kE[i];
        si[i] = kI[i];
    }
    __syncthreads();

    int idx = blockIdx.x * 256 + threadIdx.x;      // over T_DATA*SUBP
    int t = idx >> 4, s = idx & 15;
    float acc = 0.f;
    if (s < SUB_NO) {
        const float* ke = se + s * T_NO;
        const float* ki = si + s * T_NO;
        int kmax = (t + 1 < T_NO) ? (t + 1) : T_NO;
        for (int tau = 0; tau < kmax; ++tau) {
            int r = (t - tau) * SUBP + s;
            acc = fmaf(ke[tau], synE[r], acc);
            acc = fmaf(ki[tau], synI[r], acc);
        }
    }
    syn[idx] = acc;
}

// -------------------------------- Stage 4 ----------------------------------
// Single wave. lanes 0..9 own subunits. Output ring = 128 entries (>=101, so
// (t-d)&127 never hits a live slot). syn rows prefetched 64 at a time into a
// double-buffered LDS tile while the previous tile is consumed.
__global__ __launch_bounds__(32)
void scan_kernel(const float* __restrict__ syn,    // [T_DATA][SUBP]
                 const float* __restrict__ kH,     // [10][100]
                 const float* __restrict__ C_den,  // [10][10]
                 const float* __restrict__ W_sub,  // [10]
                 const float* __restrict__ V_o,    // [1]
                 const float* __restrict__ Theta,  // [10]
                 float* __restrict__ out)          // [T_DATA]
{
    __shared__ __align__(16) float r_ring[128 * SUBP];
    __shared__ __align__(16) float synbuf[2 * 64 * SUBP];

    const int lane = threadIdx.x;
    for (int i = lane; i < 128 * SUBP; i += 32) r_ring[i] = 0.f;

    // per-lane constants in registers
    float hk[T_NO];
    float wc[SUB_NO];
    float theta_s = 0.f;
    if (lane < SUB_NO) {
        #pragma unroll
        for (int d = 0; d < T_NO; ++d) hk[d] = kH[lane * T_NO + d];
        #pragma unroll
        for (int c = 0; c < SUB_NO; ++c) {
            float w = W_sub[c];
            wc[c] = w * w * C_den[lane * SUB_NO + c];
        }
        theta_s = Theta[lane];
    } else {
        #pragma unroll
        for (int d = 0; d < T_NO; ++d) hk[d] = 0.f;
        #pragma unroll
        for (int c = 0; c < SUB_NO; ++c) wc[c] = 0.f;
    }
    const float w2_0 = W_sub[0] * W_sub[0];
    const float vo   = V_o[0];

    const int CH  = 64;
    const int NCH = (T_DATA + CH - 1) / CH;

    // preload chunk 0
    float4 pre[8];
    #pragma unroll
    for (int q = 0; q < 8; ++q) {
        int idx = lane * 8 + q;
        int row = idx >> 2, col = (idx & 3) * 4;
        pre[q] = *(const float4*)(syn + (size_t)row * SUBP + col);
    }

    for (int chunk = 0; chunk < NCH; ++chunk) {
        const int bs = chunk & 1;
        // commit prefetched chunk to LDS
        #pragma unroll
        for (int q = 0; q < 8; ++q)
            ((float4*)synbuf)[bs * 256 + lane * 8 + q] = pre[q];
        __builtin_amdgcn_wave_barrier();

        // issue next chunk's loads (consumed next iteration -> overlap)
        if (chunk + 1 < NCH) {
            int nrow0 = (chunk + 1) * CH;
            #pragma unroll
            for (int q = 0; q < 8; ++q) {
                int idx = lane * 8 + q;
                int row = nrow0 + (idx >> 2), col = (idx & 3) * 4;
                if (row < T_DATA)
                    pre[q] = *(const float4*)(syn + (size_t)row * SUBP + col);
                else
                    pre[q] = make_float4(0.f, 0.f, 0.f, 0.f);
            }
        }

        const int tbase = chunk * CH;
        for (int i = 0; i < CH; ++i) {
            const int t = tbase + i;
            if (t >= T_DATA) break;
            if (lane < SUB_NO) {
                float syn_t = synbuf[bs * (64 * SUBP) + i * SUBP + lane];
                float hist = 0.f;
                #pragma unroll
                for (int d = 1; d <= T_NO; ++d)
                    hist = fmaf(r_ring[((t - d) & 127) * SUBP + lane],
                                hk[d - 1], hist);
                float prop = 0.f;
                const int ob = ((t - T_NO) & 127) * SUBP;
                #pragma unroll
                for (int c = 0; c < SUB_NO; ++c)
                    prop = fmaf(r_ring[ob + c], wc[c], prop);
                float nv = tanhf(syn_t + hist + prop + theta_s);
                r_ring[(t & 127) * SUBP + lane] = nv;
                if (lane == 0) out[t] = fmaf(nv, w2_0, vo);
            }
            __builtin_amdgcn_wave_barrier();
        }
    }
}

// ------------------------------- launcher ----------------------------------
extern "C" void kernel_launch(void* const* d_in, const int* in_sizes, int n_in,
                              void* d_out, int out_size, void* d_ws, size_t ws_size,
                              hipStream_t stream) {
    const float* S_e     = (const float*)d_in[0];
    const float* S_i     = (const float*)d_in[1];
    const float* C_den   = (const float*)d_in[2];
    const float* C_syn_e = (const float*)d_in[3];
    const float* C_syn_i = (const float*)d_in[4];
    const float* W_syn   = (const float*)d_in[5];
    const float* W_hist  = (const float*)d_in[6];
    const float* W_sub   = (const float*)d_in[7];
    const float* V_o     = (const float*)d_in[8];
    const float* Theta   = (const float*)d_in[9];
    float* out = (float*)d_out;            // [50000 final_V][3000 filters]
    float* ws  = (float*)d_ws;

    // workspace layout (floats): ~9.7 MB total
    float* kE   = ws;                      // 1000  (pad to 1024)
    float* kI   = ws + 1024;
    float* kH   = ws + 2048;
    float* synE = ws + 4096;               // 50000*16
    float* synI = synE + T_DATA * SUBP;
    float* syn  = synI + T_DATA * SUBP;

    prep_kernel<<<4, 256, 0, stream>>>(W_syn, W_hist, kE, kI, kH,
                                       out + T_DATA);

    const int ntile  = T_DATA / 16;                    // 3125
    const int nblock = (ntile + 7) / 8;                // 8 waves / block
    agg_wmma_kernel<E_NO><<<nblock, 256, 0, stream>>>(S_e, C_syn_e, synE);
    agg_wmma_kernel<I_NO><<<nblock, 256, 0, stream>>>(S_i, C_syn_i, synI);

    conv_kernel<<<(T_DATA * SUBP) / 256, 256, 0, stream>>>(synE, synI,
                                                           kE, kI, syn);

    scan_kernel<<<1, 32, 0, stream>>>(syn, kH, C_den, W_sub, V_o, Theta, out);
}
